// VGAE_44358422233714
// MI455X (gfx1250) — matrix operand order repaired and checked by reading, not compiled, not run on hardware
//
#include <hip/hip_runtime.h>
#include <hip/hip_bf16.h>

// ---------------------------------------------------------------------------
// VGAE forward for MI455X (gfx1250, wave32, WMMA).
// GEMMs: v_wmma_f32_16x16x32_bf16, fp32->bf16 on LDS stage, fragment-order
// LDS layouts so every operand fragment is 2x ds_load_b128.
// Edge aggregation: native fp32 global atomics (L2-resident accumulators).
// ---------------------------------------------------------------------------

typedef __attribute__((ext_vector_type(16))) __bf16         v16bf;
typedef __attribute__((ext_vector_type(16))) unsigned short v16us;
typedef __attribute__((ext_vector_type(8)))  float          v8f;

#define IN_CH   512
#define HID_CH  256
#define OUT_CH  64

static __device__ __forceinline__ unsigned int f2bf_pk(float a, float b) {
    union { float f; unsigned u; } ua, ub; ua.f = a; ub.f = b;
    unsigned ra = (ua.u + 0x7FFFu + ((ua.u >> 16) & 1u)) >> 16;
    unsigned rb = (ub.u + 0x7FFFu + ((ub.u >> 16) & 1u)) >> 16;
    return (ra & 0xFFFFu) | (rb << 16);
}

union frag16 { uint4 q[2]; v16us v; };

// ---------------------------------------------------------------------------
// Tiled bf16 WMMA GEMM:  C[M x ncols] = A[M x K] * W[K x ncols] (+bias)
// Block: 256 threads = 8 waves, computes a 64 x BN tile.
// Wave w: row-tile (w&3), col half (w>>2) -> BN/32 16x16 WMMA tiles.
// LDS layouts (halfword units):
//   As[r*32 + k]   : row-major, fragment reads contiguous (2x b128)
//   Bs[n*32 + k]   : column-major 32-k panel, fragment reads contiguous
// ---------------------------------------------------------------------------
template <int BN>
__global__ void __launch_bounds__(256)
vgae_gemm_bf16_wmma(const float* __restrict__ A, const float* __restrict__ W,
                    float* __restrict__ C, const float* __restrict__ bias,
                    int M, int K, int ldw, int ldc, int ccol0)
{
    constexpr int NF = BN / 32;              // fragments per wave
    __shared__ unsigned short As[64 * 32];
    __shared__ unsigned short Bs[BN * 32];

    const int tid  = threadIdx.x;
    const int wave = tid >> 5;
    const int lane = tid & 31;
    const int rt   = wave & 3;               // row tile (0..3)
    const int cp   = wave >> 2;              // col group (0..1), BN/2 cols each
    const int m0   = blockIdx.x * 64;
    const int n0   = blockIdx.y * BN;
    const int half = lane >> 4;
    const int l16  = lane & 15;

    v8f acc[NF];
#pragma unroll
    for (int f = 0; f < NF; ++f) acc[f] = (v8f){};

    // staging coordinates
    const int ar  = tid >> 2;                // A: row 0..63
    const int ac8 = (tid & 3) * 8;           // A: k chunk of 8
    for (int k0 = 0; k0 < K; k0 += 32) {
        // ---- stage A 64x32: two float4 loads -> one b128 LDS store ----
        {
            int gr = m0 + ar;
            float4 v0 = {}, v1 = {};
            if (gr < M) {
                const float4* p = (const float4*)(A + (size_t)gr * K + k0 + ac8);
                v0 = p[0]; v1 = p[1];
            }
            uint4 q;
            q.x = f2bf_pk(v0.x, v0.y); q.y = f2bf_pk(v0.z, v0.w);
            q.z = f2bf_pk(v1.x, v1.y); q.w = f2bf_pk(v1.z, v1.w);
            *(uint4*)(As + ar * 32 + ac8) = q;
        }
        // ---- stage B 32xBN swizzled: 8 coalesced b32 loads -> one b128 store
#pragma unroll
        for (int it = 0; it < (BN * 4) / 256; ++it) {
            int idx = tid + it * 256;
            int c  = idx & (BN - 1);         // column within tile
            int k8 = (idx / BN) * 8;         // k chunk of 8
            float v[8];
#pragma unroll
            for (int j = 0; j < 8; ++j)
                v[j] = W[(size_t)(k0 + k8 + j) * ldw + (n0 + c)];
            uint4 q;
            q.x = f2bf_pk(v[0], v[1]); q.y = f2bf_pk(v[2], v[3]);
            q.z = f2bf_pk(v[4], v[5]); q.w = f2bf_pk(v[6], v[7]);
            *(uint4*)(Bs + c * 32 + k8) = q;
        }
        __syncthreads();

        // ---- A fragment (16-bit A 16x32 layout): 2x b128 from LDS ----
        frag16 fa;
        {
            const char* base = (const char*)(As + (rt * 16 + l16) * 32);
            fa.q[0] = *(const uint4*)(base + half * 16);        // k 8h+0..7
            fa.q[1] = *(const uint4*)(base + 32 + half * 16);   // k 16+8h+0..7
        }
        v16bf a = __builtin_bit_cast(v16bf, fa.v);

        // ---- NF B fragments + WMMAs (A reused across all) ----
#pragma unroll
        for (int f = 0; f < NF; ++f) {
            const int bn = cp * (BN / 2) + f * 16 + l16;
            frag16 fb;
            const char* base = (const char*)(Bs + bn * 32);
            fb.q[0] = *(const uint4*)(base + half * 16);        // k 8h+0..7
            fb.q[1] = *(const uint4*)(base + 32 + half * 16);   // k 16+8h+0..7
            v16bf b = __builtin_bit_cast(v16bf, fb.v);
            acc[f] = __builtin_amdgcn_wmma_f32_16x16x32_bf16(
                false, a, false, b, (short)0, acc[f], false, false);
        }
        __syncthreads();
    }

    // ---- store C (f32 16x16 C/D layout: VGPR r -> row r + 8*half, col l16)
#pragma unroll
    for (int f = 0; f < NF; ++f) {
        const int nLoc = cp * (BN / 2) + f * 16 + l16;
        const int gcol = n0 + nLoc;
        const float bv = bias ? bias[gcol] : 0.0f;
#pragma unroll
        for (int r = 0; r < 8; ++r) {
            int gr = m0 + rt * 16 + r + half * 8;
            if (gr < M)
                C[(size_t)gr * ldc + ccol0 + gcol] = acc[f][r] + bv;
        }
    }
}

// ---------------------------------------------------------------------------
// Small helper kernels
// ---------------------------------------------------------------------------
__global__ void vgae_zero(float* __restrict__ p, long long n) {
    long long i = (long long)blockIdx.x * blockDim.x + threadIdx.x;
    if (i < n) p[i] = 0.0f;
}

__global__ void vgae_deg_init(float* __restrict__ deg, int N) {
    int i = blockIdx.x * blockDim.x + threadIdx.x;
    if (i < N) deg[i] = 1.0f;                 // self loop
}

__global__ void vgae_deg_edges(const int* __restrict__ dst, float* __restrict__ deg, int E) {
    int i = blockIdx.x * blockDim.x + threadIdx.x;
    if (i < E) unsafeAtomicAdd(&deg[dst[i]], 1.0f);
}

__global__ void vgae_dinv(const float* __restrict__ deg, float* __restrict__ dinv, int N) {
    int i = blockIdx.x * blockDim.x + threadIdx.x;
    if (i < N) dinv[i] = rsqrtf(deg[i]);      // deg >= 1 always (self loop)
}

// Edge scatter-add: agg[dst] += dinv[src]*dinv[dst] * t[src], C channels
template <int C>
__global__ void vgae_edge_agg(const float* __restrict__ t,
                              const int* __restrict__ src, const int* __restrict__ dst,
                              const float* __restrict__ dinv,
                              float* __restrict__ agg, int E)
{
    constexpr int TPE = C / 4;                // threads per edge
    long long tid = (long long)blockIdx.x * blockDim.x + threadIdx.x;
    int e  = (int)(tid / TPE);
    int c4 = (int)(tid % TPE) * 4;
    if (e >= E) return;
    int s = src[e], d = dst[e];
    float nrm = dinv[s] * dinv[d];
    const float4 v = *(const float4*)(t + (size_t)s * C + c4);
    float* o = agg + (size_t)d * C + c4;
    unsafeAtomicAdd(o + 0, nrm * v.x);
    unsafeAtomicAdd(o + 1, nrm * v.y);
    unsafeAtomicAdd(o + 2, nrm * v.z);
    unsafeAtomicAdd(o + 3, nrm * v.w);
}

// h = relu(agg + dinv^2 * t1 + b1), in place over agg (HID_CH wide)
__global__ void vgae_epi1(float* __restrict__ aggh, const float* __restrict__ t1,
                          const float* __restrict__ dinv, const float* __restrict__ b1)
{
    int i = blockIdx.x, c = threadIdx.x;
    float d2 = dinv[i] * dinv[i];
    size_t o = (size_t)i * HID_CH + c;
    float v = aggh[o] + d2 * t1[o] + b1[c];
    aggh[o] = v > 0.0f ? v : 0.0f;
}

// mu/logvar finalize + reparameterize: writes mu, logvar to d_out and z to ws
__global__ void vgae_epi2(const float* __restrict__ agg2, const float* __restrict__ t2,
                          const float* __restrict__ dinv, const float* __restrict__ bmu,
                          const float* __restrict__ blv, const float* __restrict__ eps,
                          float* __restrict__ mu_out, float* __restrict__ lv_out,
                          float* __restrict__ z)
{
    int i = blockIdx.x, c = threadIdx.x;      // c in [0,64)
    float d2 = dinv[i] * dinv[i];
    size_t o2 = (size_t)i * 128;
    float mu = agg2[o2 + c]      + d2 * t2[o2 + c]      + bmu[c];
    float lv = agg2[o2 + 64 + c] + d2 * t2[o2 + 64 + c] + blv[c];
    size_t o = (size_t)i * 64 + c;
    mu_out[o] = mu;
    lv_out[o] = lv;
    z[o] = mu + eps[o] * __expf(0.5f * lv);
}

// ---------------------------------------------------------------------------
extern "C" void kernel_launch(void* const* d_in, const int* in_sizes, int n_in,
                              void* d_out, int out_size, void* d_ws, size_t ws_size,
                              hipStream_t stream)
{
    const float* x   = (const float*)d_in[0];
    const int*   ei  = (const int*)  d_in[1];
    const float* W1  = (const float*)d_in[2];
    const float* b1  = (const float*)d_in[3];
    const float* Wmu = (const float*)d_in[4];
    const float* bmu = (const float*)d_in[5];
    const float* Wlv = (const float*)d_in[6];
    const float* blv = (const float*)d_in[7];
    const float* Wc  = (const float*)d_in[8];
    const float* bc  = (const float*)d_in[9];
    const float* eps = (const float*)d_in[10];

    const int N = in_sizes[0] / IN_CH;        // 100000
    const int E = in_sizes[1] / 2;            // 3200000
    const int* srcI = ei;
    const int* dstI = ei + E;

    float* ws = (float*)d_ws;
    float* t1   = ws;                          // N*256 (later reused: t2 + agg2)
    float* h    = ws + (size_t)N * 256;        // N*256 (agg1 -> h; later z)
    float* t2   = t1;                          // N*128
    float* agg2 = ws + (size_t)N * 128;        // N*128 (inside dead t1 tail)
    float* z    = h;                           // N*64  (h dead by then)
    float* deg  = ws + (size_t)N * 512;        // N
    float* dinv = deg + N;                     // N

    float* outp = (float*)d_out;               // N*64
    float* mup  = outp + (size_t)N * 64;       // N*64
    float* lvp  = outp + (size_t)N * 128;      // N*64

    const int TB = 256;
    const int mBlocks = (N + 63) / 64;

    // 1) degrees + dinv
    vgae_deg_init<<<(N + TB - 1) / TB, TB, 0, stream>>>(deg, N);
    vgae_deg_edges<<<(E + TB - 1) / TB, TB, 0, stream>>>(dstI, deg, E);
    vgae_dinv<<<(N + TB - 1) / TB, TB, 0, stream>>>(deg, dinv, N);

    // 2) t1 = x @ W1   (K=512, 256 cols, BN=128)
    vgae_gemm_bf16_wmma<128><<<dim3(mBlocks, HID_CH / 128), TB, 0, stream>>>(
        x, W1, t1, nullptr, N, IN_CH, HID_CH, HID_CH, 0);

    // 3) aggregate layer 1
    vgae_zero<<<(int)(((long long)N * 256 + TB - 1) / TB), TB, 0, stream>>>(h, (long long)N * 256);
    {
        long long tot = (long long)E * (HID_CH / 4);
        vgae_edge_agg<HID_CH><<<(int)((tot + TB - 1) / TB), TB, 0, stream>>>(
            t1, srcI, dstI, dinv, h, E);
    }
    vgae_epi1<<<N, HID_CH, 0, stream>>>(h, t1, dinv, b1);

    // 4) t2[:,0:64] = h @ W_mu ; t2[:,64:128] = h @ W_lv   (K=256, BN=64)
    vgae_gemm_bf16_wmma<64><<<dim3(mBlocks, 1), TB, 0, stream>>>(
        h, Wmu, t2, nullptr, N, HID_CH, OUT_CH, 128, 0);
    vgae_gemm_bf16_wmma<64><<<dim3(mBlocks, 1), TB, 0, stream>>>(
        h, Wlv, t2, nullptr, N, HID_CH, OUT_CH, 128, 64);

    // 5) aggregate layer 2 (mu|lv interleaved, 128 ch)
    vgae_zero<<<(int)(((long long)N * 128 + TB - 1) / TB), TB, 0, stream>>>(agg2, (long long)N * 128);
    {
        long long tot = (long long)E * (128 / 4);
        vgae_edge_agg<128><<<(int)((tot + TB - 1) / TB), TB, 0, stream>>>(
            t2, srcI, dstI, dinv, agg2, E);
    }

    // 6) finalize mu/logvar + reparameterize -> z
    vgae_epi2<<<N, OUT_CH, 0, stream>>>(agg2, t2, dinv, bmu, blv, eps, mup, lvp, z);

    // 7) out = z @ W_c + b_c   (K=64, BN=64)
    vgae_gemm_bf16_wmma<64><<<dim3(mBlocks, 1), TB, 0, stream>>>(
        z, Wc, outp, bc, N, OUT_CH, OUT_CH, OUT_CH, 0);
}
